// BRC_8942121910698
// MI455X (gfx1250) — matrix operand order repaired and checked by reading, not compiled, not run on hardware
//
#include <hip/hip_runtime.h>
#include <hip/hip_bf16.h>

// ---------------------------------------------------------------------------
// BRC (Bistable Recurrent Cell), 2 layers, SEQ=1024, B=128, IN=256, H=512.
//
// Strategy (MI455X / gfx1250, wave32):
//  1) Split-bf16 WMMA GEMMs for the three per-layer input projections:
//       proj = A * U^T + bias,  A:[M=T*B, K] fp32 -> (Ah, Al) bf16 split,
//       U:[512, K] fp32 -> (Wh, Wl) bf16 split,
//       proj ~= Ah*Wh + Ah*Wl + Al*Wh   (near-fp32 accuracy, 3x bf16 WMMA)
//     using v_wmma_f32_16x16x32_bf16. Weights are L2-resident (<=512KB),
//     so register tiling + direct global loads suffices (192MB L2).
//  2) The BRC scan is elementwise in (b, h): one thread owns one (b,j) and
//     loops over all 1024 timesteps privately (no barriers, no grid sync).
//     Streaming 16B/elem/step -> ~1 GB/layer -> bandwidth-bound (~46us/layer
//     at 23.3 TB/s). Next-step triple is loaded early, the one after prefetched (global_prefetch_b8).
//
// Workspace layout (~1.26 GiB): A split buffers (sized for K=512 layer),
// 3x2 weight split buffers (reused per layer), 3 proj buffers (reused),
// y0 fp32.
// ---------------------------------------------------------------------------

typedef __attribute__((ext_vector_type(16))) __bf16 v16bf;
typedef __attribute__((ext_vector_type(8)))  __bf16 v8bf;
typedef __attribute__((ext_vector_type(8)))  float  v8f;

#define SEQ_T 1024
#define BATCH 128
#define DIN   256
#define DH    512
#define MROWS (SEQ_T * BATCH)        // 131072
#define BH    (BATCH * DH)           // 65536

// ---------------------------------------------------------------------------
// fp32 -> (hi, lo) bf16 split:  x = hi + lo + O(2^-16 * x)
// ---------------------------------------------------------------------------
__global__ __launch_bounds__(256)
void brc_cvt_split(const float* __restrict__ src, __bf16* __restrict__ hi,
                   __bf16* __restrict__ lo, size_t n) {
  size_t i = ((size_t)blockIdx.x * 256 + threadIdx.x) * 4;
#pragma unroll
  for (int k = 0; k < 4; ++k) {
    size_t j = i + k;
    if (j < n) {
      float x = src[j];
      __bf16 h = (__bf16)x;
      hi[j] = h;
      lo[j] = (__bf16)(x - (float)h);
    }
  }
}

// ---------------------------------------------------------------------------
// Projection GEMM:  out[m, n] = sum_k A[m,k] * W[n,k] + bias[n]
//   A: [M, K] bf16 split (hi, lo), row-major, K multiple of 32
//   W: [512, K] bf16 split, row-major (K contiguous -> B-operand friendly)
// Block: 256 threads = 8 waves as 4(M) x 2(N). Wave tile 32x64 (2x4 WMMA
// accumulators). Block tile 128x128. grid = (M/128, 512/128, 3); z picks
// the {c, a, h} weight set.
//
// bf16 WMMA fragment layouts as loaded here (contiguous k order, same in A and B):
//   A 16x32: lane l (r=l&15, half=l>>4) holds row r, K elems
//            {half*16 + 0..7} U {half*16 + 8..15}   -> two 16B loads
//   B 32x16: lane l holds column (l&15), 16 contiguous K starting at
//            (l>>4)*16                                 -> one 32B load
//   C/D:     VGPR r: M = (l>>4)*8 + r, N = l&15
// ---------------------------------------------------------------------------
__global__ __launch_bounds__(256)
void brc_gemm_proj(const __bf16* __restrict__ Ah, const __bf16* __restrict__ Al,
                   const __bf16* __restrict__ W0h, const __bf16* __restrict__ W1h,
                   const __bf16* __restrict__ W2h,
                   const __bf16* __restrict__ W0l, const __bf16* __restrict__ W1l,
                   const __bf16* __restrict__ W2l,
                   const float* __restrict__ b0, const float* __restrict__ b1,
                   const float* __restrict__ b2,
                   float* __restrict__ o0, float* __restrict__ o1,
                   float* __restrict__ o2,
                   int K) {
  const int z = blockIdx.z;
  const __bf16* __restrict__ Wh = (z == 0) ? W0h : ((z == 1) ? W1h : W2h);
  const __bf16* __restrict__ Wl = (z == 0) ? W0l : ((z == 1) ? W1l : W2l);
  const float*  __restrict__ bias = (z == 0) ? b0 : ((z == 1) ? b1 : b2);
  float* __restrict__ out = (z == 0) ? o0 : ((z == 1) ? o1 : o2);

  const int tid  = threadIdx.x;
  const int lane = tid & 31;
  const int w    = tid >> 5;       // 0..7
  const int wm   = w >> 1;         // 0..3 (M direction)
  const int wn   = w & 1;          // 0..1 (N direction)
  const int half = lane >> 4;      // 0/1
  const int lr   = lane & 15;

  const int m0 = blockIdx.x * 128 + wm * 32;   // wave M base
  const int n0 = blockIdx.y * 128 + wn * 64;   // wave N base

  v8f acc[2][4];
#pragma unroll
  for (int mt = 0; mt < 2; ++mt)
#pragma unroll
    for (int nt = 0; nt < 4; ++nt) {
      const float bv = bias[n0 + nt * 16 + lr];   // all 8 elems share N
#pragma unroll
      for (int i = 0; i < 8; ++i) acc[mt][nt][i] = bv;
    }

  for (int k0 = 0; k0 < K; k0 += 32) {
    // --- A fragments (hi & lo) for the 2 M-tiles -------------------------
    v16bf afh[2], afl[2];
#pragma unroll
    for (int mt = 0; mt < 2; ++mt) {
      const size_t ro = (size_t)(m0 + mt * 16 + lr) * K + k0 + half * 16;
      const v8bf a0h = *(const v8bf*)(Ah + ro);
      const v8bf a1h = *(const v8bf*)(Ah + ro + 8);
      const v8bf a0l = *(const v8bf*)(Al + ro);
      const v8bf a1l = *(const v8bf*)(Al + ro + 8);
#pragma unroll
      for (int i = 0; i < 8; ++i) {
        afh[mt][i] = a0h[i]; afh[mt][i + 8] = a1h[i];
        afl[mt][i] = a0l[i]; afl[mt][i + 8] = a1l[i];
      }
    }
    // --- B fragments (hi & lo) for the 4 N-tiles (L2-resident weights) ---
    v16bf bfh[4], bfl[4];
#pragma unroll
    for (int nt = 0; nt < 4; ++nt) {
      const size_t bo = (size_t)(n0 + nt * 16 + lr) * K + k0 + half * 16;
      bfh[nt] = *(const v16bf*)(Wh + bo);
      bfl[nt] = *(const v16bf*)(Wl + bo);
    }
    // --- 3-product split-bf16 accumulate: AhWh + AhWl + AlWh -------------
#pragma unroll
    for (int mt = 0; mt < 2; ++mt)
#pragma unroll
      for (int nt = 0; nt < 4; ++nt) {
        acc[mt][nt] = __builtin_amdgcn_wmma_f32_16x16x32_bf16(
            false, afh[mt], false, bfh[nt], (short)0, acc[mt][nt], false, false);
        acc[mt][nt] = __builtin_amdgcn_wmma_f32_16x16x32_bf16(
            false, afh[mt], false, bfl[nt], (short)0, acc[mt][nt], false, false);
        acc[mt][nt] = __builtin_amdgcn_wmma_f32_16x16x32_bf16(
            false, afl[mt], false, bfh[nt], (short)0, acc[mt][nt], false, false);
      }
  }

  // --- epilogue: C/D layout -> row-major [M, 512] ------------------------
#pragma unroll
  for (int mt = 0; mt < 2; ++mt)
#pragma unroll
    for (int nt = 0; nt < 4; ++nt) {
      const int n = n0 + nt * 16 + lr;
      const int mb = m0 + mt * 16 + half * 8;
#pragma unroll
      for (int r = 0; r < 8; ++r)
        out[(size_t)(mb + r) * DH + n] = acc[mt][nt][r];
    }
}

// ---------------------------------------------------------------------------
// BRC scan: elementwise in (b, j) -> one thread per element, private loop
// over all T timesteps. Biases were folded into the GEMM, so:
//   c = sigmoid(pc + wc*h); a = 1 + tanh(pa + wa*h);
//   h = c*h + (1-c)*tanh(ph + a*h)
// Streaming, coalesced along j. Next-step triple is loaded early and the
// step after that is prefetched (speculative prefetches are dropped on
// translation failure, so no guard needed).
// ---------------------------------------------------------------------------
__global__ __launch_bounds__(256)
void brc_scan(const float* __restrict__ PC, const float* __restrict__ PA,
              const float* __restrict__ PH,
              const float* __restrict__ wc, const float* __restrict__ wa,
              const float* __restrict__ h0, float* __restrict__ Y,
              float* __restrict__ HN) {
  const int idx = blockIdx.x * 256 + threadIdx.x;   // 0 .. BH-1
  const int j = idx & (DH - 1);
  const float wcj = wc[j];
  const float waj = wa[j];
  float h = h0[idx];

  size_t o = (size_t)idx;
  float pc = PC[o], pa = PA[o], ph = PH[o];
  for (int t = 0; t < SEQ_T; ++t) {
    const size_t on = o + BH;
    float npc = 0.f, npa = 0.f, nph = 0.f;
    if (t < SEQ_T - 1) {
      npc = PC[on]; npa = PA[on]; nph = PH[on];
      __builtin_prefetch(PC + on + BH, 0, 0);
      __builtin_prefetch(PA + on + BH, 0, 0);
      __builtin_prefetch(PH + on + BH, 0, 0);
    }
    const float c = 1.0f / (1.0f + expf(-(pc + wcj * h)));
    const float a = 1.0f + tanhf(pa + waj * h);
    h = c * h + (1.0f - c) * tanhf(ph + a * h);
    Y[o] = h;
    o = on; pc = npc; pa = npa; ph = nph;
  }
  HN[idx] = h;
}

// ---------------------------------------------------------------------------
extern "C" void kernel_launch(void* const* d_in, const int* in_sizes, int n_in,
                              void* d_out, int out_size, void* d_ws, size_t ws_size,
                              hipStream_t stream) {
  const float* x   = (const float*)d_in[0];
  const float* h0  = (const float*)d_in[1];
  const float* Uc0 = (const float*)d_in[2];
  const float* wc0 = (const float*)d_in[3];
  const float* bc0 = (const float*)d_in[4];
  const float* Ua0 = (const float*)d_in[5];
  const float* wa0 = (const float*)d_in[6];
  const float* ba0 = (const float*)d_in[7];
  const float* Uh0 = (const float*)d_in[8];
  const float* bh0 = (const float*)d_in[9];
  const float* Uc1 = (const float*)d_in[10];
  const float* wc1 = (const float*)d_in[11];
  const float* bc1 = (const float*)d_in[12];
  const float* Ua1 = (const float*)d_in[13];
  const float* wa1 = (const float*)d_in[14];
  const float* ba1 = (const float*)d_in[15];
  const float* Uh1 = (const float*)d_in[16];
  const float* bh1 = (const float*)d_in[17];

  float* y1 = (float*)d_out;                         // [T, B, H]
  float* hn = y1 + (size_t)MROWS * DH;               // [2, B, H]

  // ---- workspace carve-up (256B aligned) --------------------------------
  char* ws = (char*)d_ws;
  size_t off = 0;
  auto take = [&](size_t bytes) -> char* {
    char* p = ws + off;
    off = (off + bytes + 255) & ~(size_t)255;
    return p;
  };
  const size_t aElems = (size_t)MROWS * DH;          // sized for K=512 layer
  __bf16* Ahi = (__bf16*)take(aElems * sizeof(__bf16));
  __bf16* Alo = (__bf16*)take(aElems * sizeof(__bf16));
  __bf16* Whi[3];
  __bf16* Wlo[3];
  for (int i = 0; i < 3; ++i) Whi[i] = (__bf16*)take((size_t)DH * DH * sizeof(__bf16));
  for (int i = 0; i < 3; ++i) Wlo[i] = (__bf16*)take((size_t)DH * DH * sizeof(__bf16));
  float* proj[3];
  for (int i = 0; i < 3; ++i) proj[i] = (float*)take((size_t)MROWS * DH * sizeof(float));
  float* y0 = (float*)take((size_t)MROWS * DH * sizeof(float));
  (void)ws_size;  // ~1.26 GiB used

  auto cvt = [&](const float* s, __bf16* hi, __bf16* lo, size_t n) {
    const int blocks = (int)((n + 1023) / 1024);
    brc_cvt_split<<<blocks, 256, 0, stream>>>(s, hi, lo, n);
  };

  // ======================= Layer 0 (K = 256) =============================
  cvt(x,   Ahi,    Alo,    (size_t)MROWS * DIN);
  cvt(Uc0, Whi[0], Wlo[0], (size_t)DH * DIN);
  cvt(Ua0, Whi[1], Wlo[1], (size_t)DH * DIN);
  cvt(Uh0, Whi[2], Wlo[2], (size_t)DH * DIN);

  brc_gemm_proj<<<dim3(MROWS / 128, DH / 128, 3), 256, 0, stream>>>(
      Ahi, Alo, Whi[0], Whi[1], Whi[2], Wlo[0], Wlo[1], Wlo[2],
      bc0, ba0, bh0, proj[0], proj[1], proj[2], DIN);

  brc_scan<<<BH / 256, 256, 0, stream>>>(proj[0], proj[1], proj[2],
                                         wc0, wa0, h0, y0, hn);

  // ======================= Layer 1 (K = 512) =============================
  cvt(y0,  Ahi,    Alo,    (size_t)MROWS * DH);
  cvt(Uc1, Whi[0], Wlo[0], (size_t)DH * DH);
  cvt(Ua1, Whi[1], Wlo[1], (size_t)DH * DH);
  cvt(Uh1, Whi[2], Wlo[2], (size_t)DH * DH);

  brc_gemm_proj<<<dim3(MROWS / 128, DH / 128, 3), 256, 0, stream>>>(
      Ahi, Alo, Whi[0], Whi[1], Whi[2], Wlo[0], Wlo[1], Wlo[2],
      bc1, ba1, bh1, proj[0], proj[1], proj[2], DH);

  brc_scan<<<BH / 256, 256, 0, stream>>>(proj[0], proj[1], proj[2],
                                         wc1, wa1, h0 + BH, y1, hn + BH);
}
